// Seq2Seq_63702954934384
// MI455X (gfx1250) — matrix operand order repaired and checked by reading, not compiled, not run on hardware
//
#include <hip/hip_runtime.h>

// ---------------------------------------------------------------------------
// Types for gfx1250 WMMA (wave32).
// ---------------------------------------------------------------------------
typedef __attribute__((ext_vector_type(16))) __bf16 v16bf;
typedef __attribute__((ext_vector_type(8)))  __bf16 v8bf;
typedef __attribute__((ext_vector_type(8)))  float  v8f;

// f32 -> bf16 (round to nearest even), stored as raw u16.
__device__ __forceinline__ unsigned short f2bf(float x) {
  unsigned u = __float_as_uint(x);
  unsigned r = (u + 0x7FFFu + ((u >> 16) & 1u)) >> 16;
  return (unsigned short)r;
}

// A-fragment (16-bit A 16x32, interleaved K): p = rowBase + segBase + half*8.
__device__ __forceinline__ v16bf ld_fragA(const unsigned short* p) {
  union { v16bf v; v8bf h[2]; } u;
  u.h[0] = *(const v8bf*)(p);
  u.h[1] = *(const v8bf*)(p + 16);
  return u.v;
}

// B-fragment (16-bit B 32x16, interleaved K): p = colBase + segBase + half*8.
__device__ __forceinline__ v16bf ld_fragB(const unsigned short* p) {
  union { v16bf v; v8bf h[2]; } u;
  u.h[0] = *(const v8bf*)(p);
  u.h[1] = *(const v8bf*)(p + 16);
  return u.v;
}

__device__ __forceinline__ v8f wmma_bf16(v16bf a, v16bf b, v8f c) {
  return __builtin_amdgcn_wmma_f32_16x16x32_bf16(false, a, false, b,
                                                 (short)0, c, false, false);
}

// Constant-trip, fully-unrolled K segment: Ap/Bp already offset by half*8.
template <int KSTEPS>
__device__ __forceinline__ v8f gemm_seg(const unsigned short* __restrict__ Ap,
                                        const unsigned short* __restrict__ Bp,
                                        v8f acc) {
#pragma unroll
  for (int k = 0; k < KSTEPS; ++k)
    acc = wmma_bf16(ld_fragA(Ap + k * 32), ld_fragB(Bp + k * 32), acc);
  return acc;
}

// D-tile scatter: VGPR j -> row mt*16 + j + 8*half, col n.
__device__ __forceinline__ void store_tileD(float* __restrict__ g, int ldg,
                                            int mt, int n, int half, v8f acc) {
#pragma unroll
  for (int j = 0; j < 8; ++j) g[(mt * 16 + j + 8 * half) * ldg + n] = acc[j];
}

__device__ __forceinline__ float sigmoidf_(float x) {
  return 1.0f / (1.0f + __expf(-x));
}

// ---------------------------------------------------------------------------
// Dimensions
// ---------------------------------------------------------------------------
#define BB   32
#define CC   512
#define SS   512
#define TT   128
#define VV   512
#define EH_  256
#define DH_  256

// ---------------------------------------------------------------------------
// Prep kernels: transpose input, convert & concatenate weights to bf16.
// ---------------------------------------------------------------------------
__global__ void k_xseq(const float* __restrict__ src, unsigned short* __restrict__ xseq) {
  int idx = blockIdx.x * 256 + threadIdx.x;     // [S][B][C] = 8388608
  int s = idx >> 14;
  int r = idx & 16383;
  int b = r >> 9;
  int c = r & 511;
  xseq[idx] = f2bf(src[((size_t)b * CC + c) * SS + s]);
}

// Encoder concat weights: Wcat[dir][1024][768]
//   rows   0..511 : [Wih[n] (512) | Whh[n] (256)]      -> r,z gates (gi+gh fused)
//   rows 512..767 : [Wih[n] (512) | 0]                 -> inn
//   rows 768..1023: [0 | Whh[512+(n-768)] (256)]       -> hn
__global__ void k_wcat(const float* WihF, const float* WhhF,
                       const float* WihB, const float* WhhB,
                       unsigned short* __restrict__ Wcat) {
  int rb = blockIdx.x;            // 0..2047
  int dir = rb >> 10;
  int n = rb & 1023;
  const float* Wih = dir ? WihB : WihF;
  const float* Whh = dir ? WhhB : WhhF;
  unsigned short* row = Wcat + (size_t)rb * 768;
  for (int c = threadIdx.x; c < 768; c += 256) {
    float v = 0.0f;
    if (n < 512)      v = (c < 512) ? Wih[n * 512 + c] : Whh[n * 256 + (c - 512)];
    else if (n < 768) v = (c < 512) ? Wih[n * 512 + c] : 0.0f;
    else              v = (c >= 512) ? Whh[(n - 256) * 256 + (c - 512)] : 0.0f;
    row[c] = f2bf(v);
  }
}

// Decoder GRU0 concat: dcat0[1024][896], K = x(640) | h1(256)
__global__ void k_dcat0(const float* Wih, const float* Whh, unsigned short* __restrict__ dcat) {
  int n = blockIdx.x;
  unsigned short* row = dcat + (size_t)n * 896;
  for (int c = threadIdx.x; c < 896; c += 256) {
    float v = 0.0f;
    if (n < 512)      v = (c < 640) ? Wih[n * 640 + c] : Whh[n * 256 + (c - 640)];
    else if (n < 768) v = (c < 640) ? Wih[n * 640 + c] : 0.0f;
    else              v = (c >= 640) ? Whh[(n - 256) * 256 + (c - 640)] : 0.0f;
    row[c] = f2bf(v);
  }
}

// Decoder GRU1 concat: dcat1[1024][512], K = h1new(256) | h2(256)
__global__ void k_dcat1(const float* Wih, const float* Whh, unsigned short* __restrict__ dcat) {
  int n = blockIdx.x;
  unsigned short* row = dcat + (size_t)n * 512;
  for (int c = threadIdx.x; c < 512; c += 256) {
    float v = 0.0f;
    if (n < 512)      v = (c < 256) ? Wih[n * 256 + c] : Whh[n * 256 + (c - 256)];
    else if (n < 768) v = (c < 256) ? Wih[n * 256 + c] : 0.0f;
    else              v = (c >= 256) ? Whh[(n - 256) * 256 + (c - 256)] : 0.0f;
    row[c] = f2bf(v);
  }
}

// outW/emb -> bf16; composed biases.
__global__ void k_misc(const float* outW, const float* emb,
                       const float* ebihF, const float* ebhhF,
                       const float* ebihB, const float* ebhhB,
                       const float* d0bih, const float* d0bhh,
                       const float* d1bih, const float* d1bhh,
                       unsigned short* __restrict__ outWbf,
                       unsigned short* __restrict__ embbf,
                       float* __restrict__ bcatE,
                       float* __restrict__ bdec0, float* __restrict__ bdec1) {
  int blk = blockIdx.x;
  if (blk < 512) {
    for (int c = threadIdx.x; c < 256; c += 256)
      outWbf[blk * 256 + c] = f2bf(outW[blk * 256 + c]);
  } else if (blk < 1024) {
    int e = blk - 512;
    for (int c = threadIdx.x; c < 128; c += 256)
      embbf[e * 128 + c] = f2bf(emb[e * 128 + c]);
  } else {
    const float* bih; const float* bhh; float* dst;
    if (blk == 1024)      { bih = ebihF; bhh = ebhhF; dst = bcatE;        }
    else if (blk == 1025) { bih = ebihB; bhh = ebhhB; dst = bcatE + 1024; }
    else if (blk == 1026) { bih = d0bih; bhh = d0bhh; dst = bdec0;        }
    else                  { bih = d1bih; bhh = d1bhh; dst = bdec1;        }
    for (int j = threadIdx.x; j < 1024; j += 256) {
      float v;
      if (j < 512)      v = bih[j] + bhh[j];   // r,z fused bias
      else if (j < 768) v = bih[j];            // inn
      else              v = bhh[j - 256];      // hn
      dst[j] = v;
    }
  }
}

// ---------------------------------------------------------------------------
// Encoder recurrence: one persistent workgroup per direction (grid = 2).
// Per step s: gates[32x1024] = [x_s | h] (32x768 bf16) @ Wcat^T via WMMA,
// then GRU pointwise update, h kept in LDS (f32 + bf16 copies).
// Wave-level N-group split (all control flow scalar via readfirstlane):
//   waves 0-3: nt 0..31  (r,z : K = 512 global-X  + 256 LDS-h)
//   waves 4-5: nt 32..47 (inn : K = 512 global-X)
//   waves 6-7: nt 48..63 (hn  : K = 256 LDS-h, B cols 512..767)
// ---------------------------------------------------------------------------
__global__ void __launch_bounds__(256)
k_encoder(const unsigned short* __restrict__ xseq,
          const unsigned short* __restrict__ WcatE,
          const float* __restrict__ bcatE,
          float* __restrict__ enc) {
  extern __shared__ char smem[];
  unsigned short* hbf = (unsigned short*)smem;                 // 32*256 bf16
  float* hf    = (float*)(smem + 16384);                       // 32*256 f32
  float* gates = (float*)(smem + 16384 + 32768);               // 32*1024 f32

  const int dir  = blockIdx.x;
  const unsigned short* W = WcatE + (size_t)dir * 1024 * 768;
  const float* bias = bcatE + dir * 1024;

  const int tid  = threadIdx.x;
  const int wave = __builtin_amdgcn_readfirstlane(tid >> 5);   // SGPR: scalar branches
  const int lane = tid & 31;
  const int half = lane >> 4;
  const int l15  = lane & 15;

  for (int i = tid; i < BB * EH_; i += 256) { hf[i] = 0.0f; hbf[i] = 0; }
  __syncthreads();

  for (int s = 0; s < SS; ++s) {
    const int sx = dir ? (SS - 1 - s) : s;
    const unsigned short* X = xseq + (size_t)sx * BB * CC;     // [32][512] bf16

    if (wave < 4) {
      // 16 tiles each: nt = wave*8 + (ti>>1) in 0..31, mt = ti&1
#pragma unroll 2
      for (int ti = 0; ti < 16; ++ti) {
        const int tile = wave * 16 + ti;
        const int mt = tile & 1;
        const int nt = tile >> 1;
        const int row = mt * 16 + l15;
        const unsigned short* Brow = W + (size_t)(nt * 16 + l15) * 768;
        v8f acc = {};
        acc = gemm_seg<16>(X + row * CC + half * 8, Brow + half * 8, acc);
        acc = gemm_seg<8>(hbf + row * EH_ + half * 8, Brow + 512 + half * 8, acc);
        store_tileD(gates, 1024, mt, nt * 16 + l15, half, acc);
      }
    } else if (wave < 6) {
      // waves 4,5 -> 16 tiles each covering nt 32..47
#pragma unroll 2
      for (int ti = 0; ti < 16; ++ti) {
        const int tile = (wave - 4) * 16 + ti;
        const int mt = tile & 1;
        const int nt = 32 + (tile >> 1);
        const int row = mt * 16 + l15;
        const unsigned short* Brow = W + (size_t)(nt * 16 + l15) * 768;
        v8f acc = {};
        acc = gemm_seg<16>(X + row * CC + half * 8, Brow + half * 8, acc);
        store_tileD(gates, 1024, mt, nt * 16 + l15, half, acc);
      }
    } else {
      // waves 6,7 -> 16 tiles each covering nt 48..63
#pragma unroll 2
      for (int ti = 0; ti < 16; ++ti) {
        const int tile = (wave - 6) * 16 + ti;
        const int mt = tile & 1;
        const int nt = 48 + (tile >> 1);
        const int row = mt * 16 + l15;
        const unsigned short* Brow = W + (size_t)(nt * 16 + l15) * 768;
        v8f acc = {};
        acc = gemm_seg<8>(hbf + row * EH_ + half * 8, Brow + 512 + half * 8, acc);
        store_tileD(gates, 1024, mt, nt * 16 + l15, half, acc);
      }
    }
    __syncthreads();

    // GRU pointwise: 32*256 outputs.
    for (int i = tid; i < BB * EH_; i += 256) {
      const int b = i >> 8, j = i & 255;
      const float* g = gates + b * 1024;
      float r   = sigmoidf_(g[j]       + bias[j]);
      float z   = sigmoidf_(g[256 + j] + bias[256 + j]);
      float inn = g[512 + j] + bias[512 + j];
      float hn  = g[768 + j] + bias[768 + j];
      float nv  = tanhf(inn + r * hn);
      float hnew = (1.0f - z) * nv + z * hf[i];
      hf[i]  = hnew;
      hbf[i] = f2bf(hnew);
      enc[((size_t)b * SS + sx) * 512 + dir * 256 + j] = hnew;
    }
    __syncthreads();
  }
}

// ---------------------------------------------------------------------------
// Attention enc-side score precompute: encScore[b][s] = att_b + enc[b,s,:]·attW[:512]
// ---------------------------------------------------------------------------
__global__ void k_att(const float* __restrict__ enc, const float* __restrict__ attW,
                      const float* __restrict__ attb, float* __restrict__ encScore) {
  int idx = blockIdx.x * 256 + threadIdx.x;       // 16384
  int b = idx >> 9, s = idx & 511;
  const float* er = enc + ((size_t)b * SS + s) * 512;
  float sum = attb[0];
  for (int c = 0; c < 512; ++c) sum += er[c] * attW[c];
  encScore[idx] = sum;
}

// ---------------------------------------------------------------------------
// Decoder: one persistent workgroup, 1024 threads = 32 waves (wave w <-> batch b=w).
// GEMM wave split (scalar):
//   GRU:  waves 0-15 -> nt 0..31 (K full), 16-23 -> nt 32..47 (gi K), 24-31 -> nt 48..63 (gh K)
//   OUT:  all 32 waves, 2 tiles each, K=256.
// ---------------------------------------------------------------------------
__global__ void __launch_bounds__(1024)
k_decoder(const int* __restrict__ target, const float* __restrict__ encScore,
          const float* __restrict__ enc, const float* __restrict__ attW,
          const unsigned short* __restrict__ embbf,
          const unsigned short* __restrict__ dcat0, const float* __restrict__ bdec0,
          const unsigned short* __restrict__ dcat1, const float* __restrict__ bdec1,
          const unsigned short* __restrict__ outWbf, const float* __restrict__ outb,
          float* __restrict__ out) {
  extern __shared__ char smem[];
  unsigned short* xbuf = (unsigned short*)smem;                    // 32*896 bf16: [emb|ctx|h1]
  unsigned short* buf2 = (unsigned short*)(smem + 57344);          // 32*512 bf16: [h1new|h2]
  float* h1f   = (float*)(smem + 90112);                           // 32*256
  float* h2f   = (float*)(smem + 122880);                          // 32*256
  float* gates = (float*)(smem + 155648);                          // 32*1024 (aliases wbuf)
  float* wbuf  = gates;                                            // 32*512 softmax weights
  float* hdot  = (float*)(smem + 286720);                          // 32

  const int tid  = threadIdx.x;
  const int wave = __builtin_amdgcn_readfirstlane(tid >> 5);       // SGPR
  const int lane = tid & 31;
  const int half = lane >> 4;
  const int l15  = lane & 15;
  const int b    = wave;                                           // 32 waves = 32 batches

  for (int i = tid; i < BB * DH_; i += 1024) { h1f[i] = 0.0f; h2f[i] = 0.0f; }
  __syncthreads();

  for (int t = 0; t < TT; ++t) {
    // --- phase 1: hdot[b] = h2[b,:] · attW[512:768] ---
    {
      float s = 0.0f;
#pragma unroll
      for (int jj = 0; jj < 8; ++jj) s += h2f[b * 256 + lane + 32 * jj] * attW[512 + lane + 32 * jj];
      for (int o = 16; o; o >>= 1) s += __shfl_xor(s, o, 32);
      if (lane == 0) hdot[b] = s;
    }
    __syncthreads();

    // --- phase 2: softmax over S into wbuf (wave-local) ---
    {
      float hd = hdot[b];
      float raw[16];
      float mx = -3.4e38f;
#pragma unroll
      for (int i = 0; i < 16; ++i) {
        float v = encScore[b * 512 + lane + 32 * i] + hd;
        raw[i] = v; mx = fmaxf(mx, v);
      }
      for (int o = 16; o; o >>= 1) mx = fmaxf(mx, __shfl_xor(mx, o, 32));
      float sum = 0.0f;
#pragma unroll
      for (int i = 0; i < 16; ++i) { float e = __expf(raw[i] - mx); raw[i] = e; sum += e; }
      for (int o = 16; o; o >>= 1) sum += __shfl_xor(sum, o, 32);
      float inv = 1.0f / sum;
#pragma unroll
      for (int i = 0; i < 16; ++i) wbuf[b * 512 + lane + 32 * i] = raw[i] * inv;
    }

    // --- phase 3: ctx[b,:] = sum_s w*enc ; build xbuf = [emb(tok) | ctx | h1] ---
    {
      float acc[16];
#pragma unroll
      for (int i = 0; i < 16; ++i) acc[i] = 0.0f;
      for (int s = 0; s < SS; ++s) {
        float wv = wbuf[b * 512 + s];
        const float* er = enc + ((size_t)b * SS + s) * 512;
#pragma unroll
        for (int i = 0; i < 16; ++i) acc[i] += wv * er[lane + 32 * i];
      }
#pragma unroll
      for (int i = 0; i < 16; ++i) xbuf[b * 896 + 128 + lane + 32 * i] = f2bf(acc[i]);
      int tok = target[b * TT + t];
#pragma unroll
      for (int cc = 0; cc < 4; ++cc) xbuf[b * 896 + lane + 32 * cc] = embbf[tok * 128 + lane + 32 * cc];
#pragma unroll
      for (int cc = 0; cc < 8; ++cc)
        xbuf[b * 896 + 640 + lane + 32 * cc] = f2bf(h1f[b * 256 + lane + 32 * cc]);
    }
    __syncthreads();

    // --- phase 4: GRU0 GEMM [32x1024] = xbuf(32x896) @ dcat0^T ---
    if (wave < 16) {
#pragma unroll
      for (int ti = 0; ti < 4; ++ti) {
        const int tile = wave * 4 + ti;
        const int mt = tile & 1;
        const int nt = tile >> 1;                       // 0..31
        const unsigned short* Ar = xbuf + (mt * 16 + l15) * 896;
        const unsigned short* Br = dcat0 + (size_t)(nt * 16 + l15) * 896;
        v8f acc = {};
        acc = gemm_seg<28>(Ar + half * 8, Br + half * 8, acc);
        store_tileD(gates, 1024, mt, nt * 16 + l15, half, acc);
      }
    } else if (wave < 24) {
#pragma unroll
      for (int ti = 0; ti < 4; ++ti) {
        const int tile = (wave - 16) * 4 + ti;
        const int mt = tile & 1;
        const int nt = 32 + (tile >> 1);                // 32..47
        const unsigned short* Ar = xbuf + (mt * 16 + l15) * 896;
        const unsigned short* Br = dcat0 + (size_t)(nt * 16 + l15) * 896;
        v8f acc = {};
        acc = gemm_seg<20>(Ar + half * 8, Br + half * 8, acc);
        store_tileD(gates, 1024, mt, nt * 16 + l15, half, acc);
      }
    } else {
#pragma unroll
      for (int ti = 0; ti < 4; ++ti) {
        const int tile = (wave - 24) * 4 + ti;
        const int mt = tile & 1;
        const int nt = 48 + (tile >> 1);                // 48..63
        const unsigned short* Ar = xbuf + (mt * 16 + l15) * 896;
        const unsigned short* Br = dcat0 + (size_t)(nt * 16 + l15) * 896;
        v8f acc = {};
        acc = gemm_seg<8>(Ar + 640 + half * 8, Br + 640 + half * 8, acc);
        store_tileD(gates, 1024, mt, nt * 16 + l15, half, acc);
      }
    }
    __syncthreads();

    // --- phase 5: h1 update; fill buf2 = [h1new | h2] bf16 ---
    for (int i = tid; i < BB * DH_; i += 1024) {
      const int bb = i >> 8, j = i & 255;
      const float* g = gates + bb * 1024;
      float r   = sigmoidf_(g[j]       + bdec0[j]);
      float z   = sigmoidf_(g[256 + j] + bdec0[256 + j]);
      float nv  = tanhf(g[512 + j] + bdec0[512 + j] + r * (g[768 + j] + bdec0[768 + j]));
      float h1n = (1.0f - z) * nv + z * h1f[i];
      h1f[i] = h1n;
      buf2[bb * 512 + j]       = f2bf(h1n);
      buf2[bb * 512 + 256 + j] = f2bf(h2f[i]);
    }
    __syncthreads();

    // --- phase 6: GRU1 GEMM [32x1024] = buf2(32x512) @ dcat1^T ---
    if (wave < 16) {
#pragma unroll
      for (int ti = 0; ti < 4; ++ti) {
        const int tile = wave * 4 + ti;
        const int mt = tile & 1;
        const int nt = tile >> 1;
        const unsigned short* Ar = buf2 + (mt * 16 + l15) * 512;
        const unsigned short* Br = dcat1 + (size_t)(nt * 16 + l15) * 512;
        v8f acc = {};
        acc = gemm_seg<16>(Ar + half * 8, Br + half * 8, acc);
        store_tileD(gates, 1024, mt, nt * 16 + l15, half, acc);
      }
    } else if (wave < 24) {
#pragma unroll
      for (int ti = 0; ti < 4; ++ti) {
        const int tile = (wave - 16) * 4 + ti;
        const int mt = tile & 1;
        const int nt = 32 + (tile >> 1);
        const unsigned short* Ar = buf2 + (mt * 16 + l15) * 512;
        const unsigned short* Br = dcat1 + (size_t)(nt * 16 + l15) * 512;
        v8f acc = {};
        acc = gemm_seg<8>(Ar + half * 8, Br + half * 8, acc);
        store_tileD(gates, 1024, mt, nt * 16 + l15, half, acc);
      }
    } else {
#pragma unroll
      for (int ti = 0; ti < 4; ++ti) {
        const int tile = (wave - 24) * 4 + ti;
        const int mt = tile & 1;
        const int nt = 48 + (tile >> 1);
        const unsigned short* Ar = buf2 + (mt * 16 + l15) * 512;
        const unsigned short* Br = dcat1 + (size_t)(nt * 16 + l15) * 512;
        v8f acc = {};
        acc = gemm_seg<8>(Ar + 256 + half * 8, Br + 256 + half * 8, acc);
        store_tileD(gates, 1024, mt, nt * 16 + l15, half, acc);
      }
    }
    __syncthreads();

    // --- phase 7: h2 update; write h2new bf16 into buf2[:, 0:256] ---
    for (int i = tid; i < BB * DH_; i += 1024) {
      const int bb = i >> 8, j = i & 255;
      const float* g = gates + bb * 1024;
      float r   = sigmoidf_(g[j]       + bdec1[j]);
      float z   = sigmoidf_(g[256 + j] + bdec1[256 + j]);
      float nv  = tanhf(g[512 + j] + bdec1[512 + j] + r * (g[768 + j] + bdec1[768 + j]));
      float h2n = (1.0f - z) * nv + z * h2f[i];
      h2f[i] = h2n;
      buf2[bb * 512 + j] = f2bf(h2n);
    }
    __syncthreads();

    // --- phase 8: logits = h2new(32x256) @ outW^T + outb -> d_out[b][t][v] ---
#pragma unroll
    for (int ti = 0; ti < 2; ++ti) {
      const int tile = wave * 2 + ti;
      const int mt = tile & 1;
      const int vt = tile >> 1;                     // 0..31
      const unsigned short* Ar = buf2 + (mt * 16 + l15) * 512;
      const unsigned short* Br = outWbf + (size_t)(vt * 16 + l15) * 256;
      v8f acc = {};
      acc = gemm_seg<8>(Ar + half * 8, Br + half * 8, acc);
      const int v = vt * 16 + l15;
      const float biasv = outb[v];
#pragma unroll
      for (int j = 0; j < 8; ++j) {
        const int bb = mt * 16 + j + 8 * half;
        out[((size_t)bb * TT + t) * VV + v] = acc[j] + biasv;
      }
    }
    __syncthreads();
  }
}

// ---------------------------------------------------------------------------
// Host launcher
// ---------------------------------------------------------------------------
extern "C" void kernel_launch(void* const* d_in, const int* in_sizes, int n_in,
                              void* d_out, int out_size, void* d_ws, size_t ws_size,
                              hipStream_t stream) {
  const float* src      = (const float*)d_in[0];
  const int*   target   = (const int*)d_in[1];
  const float* eWihF    = (const float*)d_in[2];
  const float* eWhhF    = (const float*)d_in[3];
  const float* ebihF    = (const float*)d_in[4];
  const float* ebhhF    = (const float*)d_in[5];
  const float* eWihB    = (const float*)d_in[6];
  const float* eWhhB    = (const float*)d_in[7];
  const float* ebihB    = (const float*)d_in[8];
  const float* ebhhB    = (const float*)d_in[9];
  const float* attW     = (const float*)d_in[10];
  const float* attb     = (const float*)d_in[11];
  const float* emb      = (const float*)d_in[12];
  const float* d0Wih    = (const float*)d_in[13];
  const float* d0Whh    = (const float*)d_in[14];
  const float* d0bih    = (const float*)d_in[15];
  const float* d0bhh    = (const float*)d_in[16];
  const float* d1Wih    = (const float*)d_in[17];
  const float* d1Whh    = (const float*)d_in[18];
  const float* d1bih    = (const float*)d_in[19];
  const float* d1bhh    = (const float*)d_in[20];
  const float* outW     = (const float*)d_in[21];
  const float* outb     = (const float*)d_in[22];
  float* out = (float*)d_out;

  char* ws = (char*)d_ws;
  size_t off = 0;
  auto carve = [&](size_t bytes) -> char* {
    char* p = ws + off;
    off = (off + bytes + 255) & ~(size_t)255;
    return p;
  };
  unsigned short* xseq   = (unsigned short*)carve((size_t)SS * BB * CC * 2);    // 16 MB
  unsigned short* WcatE  = (unsigned short*)carve((size_t)2 * 1024 * 768 * 2);  // 3 MB
  float*          bcatE  = (float*)carve(2 * 1024 * 4);
  unsigned short* dcat0  = (unsigned short*)carve((size_t)1024 * 896 * 2);
  float*          bdec0  = (float*)carve(1024 * 4);
  unsigned short* dcat1  = (unsigned short*)carve((size_t)1024 * 512 * 2);
  float*          bdec1  = (float*)carve(1024 * 4);
  unsigned short* outWbf = (unsigned short*)carve((size_t)512 * 256 * 2);
  unsigned short* embbf  = (unsigned short*)carve((size_t)512 * 128 * 2);
  float*          enc    = (float*)carve((size_t)BB * SS * 512 * 4);            // 32 MB
  float*          encSc  = (float*)carve((size_t)BB * SS * 4);
  (void)in_sizes; (void)n_in; (void)out_size; (void)ws_size;

  // Prep (parallel, grid-wide)
  k_xseq <<<32768, 256, 0, stream>>>(src, xseq);
  k_wcat <<<2048, 256, 0, stream>>>(eWihF, eWhhF, eWihB, eWhhB, WcatE);
  k_dcat0<<<1024, 256, 0, stream>>>(d0Wih, d0Whh, dcat0);
  k_dcat1<<<1024, 256, 0, stream>>>(d1Wih, d1Whh, dcat1);
  k_misc <<<1028, 256, 0, stream>>>(outW, emb, ebihF, ebhhF, ebihB, ebhhB,
                                    d0bih, d0bhh, d1bih, d1bhh,
                                    outWbf, embbf, bcatE, bdec0, bdec1);

  // Bidirectional encoder recurrence: 1 persistent WGP per direction.
  k_encoder<<<2, 256, 180224, stream>>>(xseq, WcatE, bcatE, enc);

  // Attention enc-side scores (parallel).
  k_att<<<64, 256, 0, stream>>>(enc, attW, attb, encSc);

  // Decoder: 1 persistent workgroup, 32 waves.
  k_decoder<<<1, 1024, 286976, stream>>>(target, encSc, enc, attW, embbf,
                                         dcat0, bdec0, dcat1, bdec1,
                                         outWbf, outb, out);
}
